// HomoAttention_35450660061721
// MI455X (gfx1250) — matrix operand order repaired and checked by reading, not compiled, not run on hardware
//
#include <hip/hip_runtime.h>

#define NB  64
#define NF  8
#define ED  32
#define DD  256          // NF*ED
#define WPAD 72          // padded row stride (halves) of weight matrix: bank-conflict-free, 16B aligned
#define EPSN 1e-12f

typedef __attribute__((ext_vector_type(16))) _Float16     v16h;
typedef __attribute__((ext_vector_type(8)))  _Float16     v8h;
typedef __attribute__((ext_vector_type(8)))  float        v8f;
typedef __attribute__((ext_vector_type(4)))  unsigned int v4u;
typedef __attribute__((ext_vector_type(8)))  int          v8i;
typedef __attribute__((ext_vector_type(4)))  int          v4i;

// dynamic LDS partition (bytes)
#define OFF_Z    0                        // 65536  f32 z tile [64][256]
#define OFF_XHAT 65536                    // 1024   f32 normalized features
#define OFF_A    66560                    // 2048   f32 logits  [f][n]
#define OFF_RN   68608                    // 2048   f32 1/||z|| [f][n]
#define OFF_W    70656                    // 2304   f16 weights [16][WPAD] (rows 8..15 zero)
#define SMEM_BYTES 72960

__global__ __launch_bounds__(256, 1) void homo_attn_kernel(
    const float* __restrict__ feat,      // [B, 256]
    const float* __restrict__ neigh,     // [B, 64, 256]
    float* __restrict__ out)             // [B, 256]
{
    extern __shared__ char smem[];
    float*    s_z    = (float*)(smem + OFF_Z);
    float*    s_xhat = (float*)(smem + OFF_XHAT);
    float*    s_a    = (float*)(smem + OFF_A);
    float*    s_rn   = (float*)(smem + OFF_RN);
    _Float16* s_w    = (_Float16*)(smem + OFF_W);

    const int tid  = threadIdx.x;
    const int lane = tid & 31;
    const int wid  = tid >> 5;            // wave id 0..7 == facet id
    const size_t b = blockIdx.x;

    // ---------- stage 0a: one TDM descriptor streams the whole 64KB z tile into LDS ----------
    if (wid == 0) {
        uint64_t ga  = (uint64_t)(uintptr_t)(neigh + b * (size_t)(NB * DD));
        uint32_t lds = (uint32_t)(uintptr_t)s_z;      // low 32 bits = LDS byte offset
        // D# group 0: count=1 | lds_addr | global_addr[56:0] | type=2
        v4u g0 = { 1u,
                   lds,
                   (uint32_t)ga,
                   (uint32_t)((ga >> 32) & 0x01FFFFFFu) | (2u << 30) };
        // D# group 1: data_size=4B; tensor 256 x 64, tile 256 x 64, dim0 stride 256
        v8i g1;
        g1[0] = (int)(2u << 16);              // workgroup_mask=0 | data_size=2 (4B)
        g1[1] = (int)(256u << 16);            // tensor_dim0[15:0]=256 in [31:16]
        g1[2] = (int)(64u  << 16);            // tensor_dim0[31:16]=0 | tensor_dim1[15:0]=64
        g1[3] = (int)(256u << 16);            // tensor_dim1[31:16]=0 | tile_dim0=256
        g1[4] = (int)(64u);                   // tile_dim1=64 | tile_dim2=0 (unused)
        g1[5] = (int)(256);                   // tensor_dim0_stride[31:0]=256
        g1[6] = (int)(16384u << 16);          // stride0[47:32]=0 | tensor_dim1_stride[15:0]
        g1[7] = 0;                            // tensor_dim1_stride[47:16]
        v4i gz4 = { 0, 0, 0, 0 };             // groups 2/3 unused (<=2D tile)
        v8i gz8 = { 0, 0, 0, 0, 0, 0, 0, 0 }; // extra SGPR group (unused), 6-arg toolchain form
        __builtin_amdgcn_tensor_load_to_lds(g0, g1, gz4, gz4, gz8, 0);
    }

    // ---------- stage 0b (overlapped with DMA): zero W pad rows, normalize x ----------
    #pragma unroll
    for (int i = 0; i < 2; ++i) {
        int k = tid + 256 * i;                        // 0..511 -> rows m=8..15, n=0..63
        s_w[(8 + (k >> 6)) * WPAD + (k & 63)] = (_Float16)0.0f;
    }
    {
        float xv = feat[b * (size_t)DD + tid];        // wave f owns facet f
        float ss = xv * xv;
        #pragma unroll
        for (int off = 16; off > 0; off >>= 1) ss += __shfl_xor(ss, off, 32);
        s_xhat[tid] = xv / fmaxf(sqrtf(ss), EPSN);
    }
    if (wid == 0) __builtin_amdgcn_s_wait_tensorcnt(0);
    __syncthreads();

    // ---------- stage 1: logits + inverse norms for all 512 (n,f) pairs (full f32) ----------
    #pragma unroll
    for (int i = 0; i < 2; ++i) {
        int p = tid + 256 * i;                        // 0..511
        int n = p >> 3, f = p & 7;
        const float* zp = s_z + n * DD + f * ED;
        const float* xp = s_xhat + f * ED;
        float dot = 0.f, ss = 0.f;
        #pragma unroll
        for (int e = 0; e < ED; ++e) {
            float zv = zp[e];
            dot = fmaf(zv, xp[e], dot);
            ss  = fmaf(zv, zv, ss);
        }
        float rn = 1.0f / fmaxf(sqrtf(ss), EPSN);
        s_a [f * NB + n] = dot * rn;                  // cosine logit
        s_rn[f * NB + n] = rn;
    }
    __syncthreads();

    // ---------- stage 2: softmax over 64 neighbors; one wave32 per facet ----------
    {
        int f = wid;
        float a0 = s_a[f * NB + lane];
        float a1 = s_a[f * NB + lane + 32];
        float m = fmaxf(a0, a1);
        #pragma unroll
        for (int off = 16; off > 0; off >>= 1) m = fmaxf(m, __shfl_xor(m, off, 32));
        float e0 = __expf(a0 - m), e1 = __expf(a1 - m);
        float s = e0 + e1;
        #pragma unroll
        for (int off = 16; off > 0; off >>= 1) s += __shfl_xor(s, off, 32);
        float inv = 1.0f / s;
        // fold 1/||z|| into the weight; store W^T row-major [m][n] for b128 A-fragment loads
        s_w[f * WPAD + lane]      = (_Float16)(e0 * inv * s_rn[f * NB + lane]);
        s_w[f * WPAD + lane + 32] = (_Float16)(e1 * inv * s_rn[f * NB + lane + 32]);
    }
    __syncthreads();

    // ---------- stage 3: U(16x256) = W^T(16x64) x Z(64x256) via v_wmma_f32_16x16x32_f16 ----
    // Each wave: 2 N-tiles (t = 2*wid, 2*wid+1), K = 64 in two 32-steps.
    // Tile t's 16 columns lie inside facet block t/2 == wid -> keep D row f = wid.
    {
        const int m  = lane & 15;                     // A row (facet), B column within tile
        const int hi = lane >> 4;                     // lane-half selector
        const v8h* wrow = (const v8h*)(s_w + m * WPAD);
        #pragma unroll
        for (int i = 0; i < 2; ++i) {
            const int t = 2 * wid + i;                // columns 16t .. 16t+15
            v8f acc = {};
            #pragma unroll
            for (int s = 0; s < 2; ++s) {
                // A (16x32 f16): lane's K indices = {32s+16hi+0..15}
                v8h alo = wrow[4 * s + 2 * hi];
                v8h ahi = wrow[4 * s + 2 * hi + 1];
                v16h A = __builtin_shufflevector(alo, ahi,
                         0,1,2,3,4,5,6,7,8,9,10,11,12,13,14,15);
                // B (32x16 f16): K = j + 16*hi, N = lane%16; f32 -> f16 on read
                v16h B;
                #pragma unroll
                for (int j = 0; j < 16; ++j) {
                    int kb = j + hi * 16;
                    B[j] = (_Float16)s_z[(32 * s + kb) * DD + 16 * t + m];
                }
                acc = __builtin_amdgcn_wmma_f32_16x16x32_f16(
                    /*neg_a=*/false, A, /*neg_b=*/false, B,
                    /*c_mod=*/(short)0, acc, /*reuse_a=*/false, /*reuse_b=*/false);
            }
            // D row f=wid lives in acc[f] on lanes 0..15 (N = lane)
            const int f = wid;
            float val = 0.f;
            #pragma unroll
            for (int v = 0; v < 8; ++v) val = (v == f) ? acc[v] : val;
            if (lane < 16)
                out[b * (size_t)DD + 16 * t + lane] = val;
        }
    }
}

extern "C" void kernel_launch(void* const* d_in, const int* in_sizes, int n_in,
                              void* d_out, int out_size, void* d_ws, size_t ws_size,
                              hipStream_t stream) {
    const float* feat  = (const float*)d_in[0];   // [B, 256] f32
    const float* neigh = (const float*)d_in[1];   // [B, 64, 256] f32
    float* out = (float*)d_out;                   // [B, 256] f32
    const int B = in_sizes[0] / DD;               // 8192
    homo_attn_kernel<<<dim3(B), dim3(256), SMEM_BYTES, stream>>>(feat, neigh, out);
}
